// MambaLayer_82480551952904
// MI455X (gfx1250) — matrix operand order repaired and checked by reading, not compiled, not run on hardware
//
#include <hip/hip_runtime.h>

typedef __attribute__((ext_vector_type(16))) _Float16 v16h;
typedef __attribute__((ext_vector_type(8)))  float    v8f;

#define B_SZ    4
#define L_SEQ   2048
#define DIM_    512
#define D_STATE 16
#define D_CONV  4
#define D_INNER 1024
#define DT_RANK 32
#define NTOK    (B_SZ * L_SEQ)   /* 8192 */

// ---------------------------------------------------------------------------
// f32 -> f16 conversion (activations, row-major preserved)
// ---------------------------------------------------------------------------
__global__ void cvt_f32_f16(const float* __restrict__ src,
                            _Float16* __restrict__ dst, int n) {
  int i = blockIdx.x * blockDim.x + threadIdx.x;
  if (i < n) dst[i] = (_Float16)src[i];
}

// f32 (R x C, row-major) -> f16 transposed (C x R): weights staged once per
// call so the GEMM's B operand is column-major and LDS staging is all-b128.
__global__ void cvt_transpose_f16(const float* __restrict__ src,
                                  _Float16* __restrict__ dst, int R, int C) {
  int i = blockIdx.x * blockDim.x + threadIdx.x;
  if (i >= R * C) return;
  const int r = i / C, c = i - r * C;
  dst[(size_t)c * R + r] = (_Float16)src[i];
}

// ---------------------------------------------------------------------------
// f16 GEMM: C(MxN) = A(MxK, row-major) * B (given as B^T, NxK column-major
// weights). v_wmma_f32_16x16x32_f16. Block tile 128x64, 8 waves (wave32) in a
// 4x2 grid; each wave owns a 32x32 output tile = 4 WMMA accumulators fed by
// 2 A-frags + 2 B-frags per K-step (4 wmma : 8 ds_load_b128; staging is
// 3 ds_store_b128/thread). Optional fused softplus(+bias) epilogue.
// Requires M%128==0, N%64==0, K%32==0.
// ---------------------------------------------------------------------------
__global__ __launch_bounds__(256)
void wmma_gemm_f16f32(const _Float16* __restrict__ A,
                      const _Float16* __restrict__ BT,   // [N][K]
                      float* __restrict__ C,
                      const float* __restrict__ bias, int softplus_act,
                      int M, int N, int K, int lda, int ldc) {
  __shared__ _Float16 As[128][40];  // 128 rows x 32 K (padded, 10.0 KB)
  __shared__ _Float16 Bt[64][40];   // 64 cols x 32 K (padded, 5.0 KB)

  const int tid   = threadIdx.x;
  const int lane  = tid & 31;
  const int wid   = tid >> 5;
  const int waveM = wid >> 1;            // 0..3 -> 32-row strip
  const int waveN = wid & 1;             // 0..1 -> 32-col strip
  const int m0 = blockIdx.y * 128;
  const int n0 = blockIdx.x * 64;
  const int hi  = lane >> 4;             // lane-half select (K split per ISA layout)
  const int r16 = lane & 15;

  const int ar = tid >> 2, ac = (tid & 3) * 8;   // staging rows ar, ar+64 / Bt row ar

  v8f c00 = {}, c01 = {}, c10 = {}, c11 = {};

  for (int k0 = 0; k0 < K; k0 += 32) {
    if (k0 + 32 < K) {
      __builtin_prefetch(&A[(size_t)(m0 + ar) * lda + k0 + 32 + ac], 0, 1);
      __builtin_prefetch(&A[(size_t)(m0 + ar + 64) * lda + k0 + 32 + ac], 0, 1);
      __builtin_prefetch(&BT[(size_t)(n0 + ar) * K + k0 + 32 + ac], 0, 1);
    }
    // stage A tile (two rows per thread) and B^T tile — all 16B vector copies
    *reinterpret_cast<uint4*>(&As[ar][ac]) =
        *reinterpret_cast<const uint4*>(&A[(size_t)(m0 + ar) * lda + k0 + ac]);
    *reinterpret_cast<uint4*>(&As[ar + 64][ac]) =
        *reinterpret_cast<const uint4*>(&A[(size_t)(m0 + ar + 64) * lda + k0 + ac]);
    *reinterpret_cast<uint4*>(&Bt[ar][ac]) =
        *reinterpret_cast<const uint4*>(&BT[(size_t)(n0 + ar) * K + k0 + ac]);
    __syncthreads();

    // A fragments 16x32: lanes 0-15 K {0..7,16..23}, lanes 16-31 K {8..15,24..31}
    v16h af0, af1;
    const int rowA = waveM * 32 + r16;
#pragma unroll
    for (int i = 0; i < 16; ++i) {
      const int v = i >> 1, h = i & 1;
      const int kk = ((v & 4) ? 16 : 0) + hi * 8 + (v & 3) * 2 + h;
      af0[i] = As[rowA][kk];
      af1[i] = As[rowA + 16][kk];
    }
    // B fragments 32x16: lanes 0-15 K {0..7,16..23}, lanes 16-31 K {8..15,24..31}
    v16h bf0, bf1;
    const int colB = waveN * 32 + r16;
#pragma unroll
    for (int i = 0; i < 16; ++i) {
      const int kk = ((i & 8) ? 16 : 0) + hi * 8 + (i & 7);
      bf0[i] = Bt[colB][kk];
      bf1[i] = Bt[colB + 16][kk];
    }
    c00 = __builtin_amdgcn_wmma_f32_16x16x32_f16(false, af0, false, bf0,
                                                 (short)0, c00, false, false);
    c01 = __builtin_amdgcn_wmma_f32_16x16x32_f16(false, af0, false, bf1,
                                                 (short)0, c01, false, false);
    c10 = __builtin_amdgcn_wmma_f32_16x16x32_f16(false, af1, false, bf0,
                                                 (short)0, c10, false, false);
    c11 = __builtin_amdgcn_wmma_f32_16x16x32_f16(false, af1, false, bf1,
                                                 (short)0, c11, false, false);
    __syncthreads();
  }

  // D layout: lane 0-15 -> M = v, lane 16-31 -> M = 8+v; N = lane&15
  const int row0 = m0 + waveM * 32 + hi * 8;
  const int col0 = n0 + waveN * 32 + r16;
  const float b0 = (softplus_act != 0) ? bias[col0]      : 0.0f;
  const float b1 = (softplus_act != 0) ? bias[col0 + 16] : 0.0f;
#pragma unroll
  for (int v = 0; v < 8; ++v) {
    float e00 = c00[v], e01 = c01[v], e10 = c10[v], e11 = c11[v];
    if (softplus_act != 0) {
      e00 += b0; e00 = (e00 > 20.0f) ? e00 : __logf(1.0f + __expf(e00));
      e01 += b1; e01 = (e01 > 20.0f) ? e01 : __logf(1.0f + __expf(e01));
      e10 += b0; e10 = (e10 > 20.0f) ? e10 : __logf(1.0f + __expf(e10));
      e11 += b1; e11 = (e11 > 20.0f) ? e11 : __logf(1.0f + __expf(e11));
    }
    C[(size_t)(row0 + v) * ldc + col0]           = e00;
    C[(size_t)(row0 + v) * ldc + col0 + 16]      = e01;
    C[(size_t)(row0 + 16 + v) * ldc + col0]      = e10;
    C[(size_t)(row0 + 16 + v) * ldc + col0 + 16] = e11;
  }
}

// ---------------------------------------------------------------------------
// Causal depthwise conv1d (k=4) + bias + SiLU over the xh half of xz.
// ---------------------------------------------------------------------------
__global__ void conv_silu_kernel(const float* __restrict__ xz,
                                 const float* __restrict__ conv_w,
                                 const float* __restrict__ conv_b,
                                 float* __restrict__ xh32,
                                 _Float16* __restrict__ xh16) {
  int idx = blockIdx.x * blockDim.x + threadIdx.x;
  if (idx >= NTOK * D_INNER) return;
  const int e   = idx & (D_INNER - 1);
  const int tok = idx >> 10;
  const int l   = tok & (L_SEQ - 1);
  const int b   = tok >> 11;
  float acc = conv_b[e];
#pragma unroll
  for (int j = 0; j < D_CONV; ++j) {
    const int ls = l - (D_CONV - 1) + j;
    if (ls >= 0)
      acc += conv_w[e * D_CONV + j] *
             xz[((size_t)b * L_SEQ + ls) * (2 * D_INNER) + e];
  }
  const float s = acc / (1.0f + __expf(-acc));   // SiLU
  xh32[idx] = s;
  xh16[idx] = (_Float16)s;
}

// dbl[:, :32] -> f16 dtl for the delta GEMM
__global__ void dtl_cvt_kernel(const float* __restrict__ dbl,
                               _Float16* __restrict__ dtl) {
  int i = blockIdx.x * blockDim.x + threadIdx.x;
  if (i >= NTOK * DT_RANK) return;
  const int row = i >> 5, j = i & 31;
  dtl[i] = (_Float16)dbl[(size_t)row * (DT_RANK + 2 * D_STATE) + j];
}

// ---------------------------------------------------------------------------
// Selective scan: sequential over L, parallel over (b, e). 64 lanes/block,
// 64 blocks -> 4096 channels. 16 states/lane in registers; B_t/C_t staged
// in LDS once per timestep. Fuses the D_skip residual.
// ---------------------------------------------------------------------------
__global__ __launch_bounds__(64)
void scan_kernel(const float* __restrict__ delta,
                 const float* __restrict__ dbl,
                 const float* __restrict__ xh32,
                 const float* __restrict__ A_log,
                 const float* __restrict__ D_skip,
                 float* __restrict__ yssm) {
  __shared__ float sBC[2 * D_STATE];
  const int b = blockIdx.x >> 4;
  const int e = ((blockIdx.x & 15) << 6) + threadIdx.x;

  float Arow[D_STATE], h[D_STATE];
#pragma unroll
  for (int n = 0; n < D_STATE; ++n) {
    Arow[n] = -__expf(A_log[e * D_STATE + n]);
    h[n] = 0.0f;
  }
  const float dsk = D_skip[e];

  for (int t = 0; t < L_SEQ; ++t) {
    const size_t tok = (size_t)b * L_SEQ + t;
    if (threadIdx.x < 2 * D_STATE)
      sBC[threadIdx.x] = dbl[tok * (DT_RANK + 2 * D_STATE) + DT_RANK + threadIdx.x];
    __syncthreads();
    const float d  = delta[tok * D_INNER + e];
    const float xv = xh32[tok * D_INNER + e];
    const float dx = d * xv;
    float y = 0.0f;
#pragma unroll
    for (int n = 0; n < D_STATE; ++n) {
      h[n] = h[n] * __expf(d * Arow[n]) + dx * sBC[n];
      y += h[n] * sBC[D_STATE + n];
    }
    yssm[tok * D_INNER + e] = y + xv * dsk;
    __syncthreads();
  }
}

// yg = (scan output) * silu(z), emitted as f16 for the final WMMA GEMM
__global__ void gate_kernel(const float* __restrict__ yssm,
                            const float* __restrict__ xz,
                            _Float16* __restrict__ yg) {
  int i = blockIdx.x * blockDim.x + threadIdx.x;
  if (i >= NTOK * D_INNER) return;
  const int e = i & (D_INNER - 1), row = i >> 10;
  const float z = xz[(size_t)row * (2 * D_INNER) + D_INNER + e];
  const float g = z / (1.0f + __expf(-z));
  yg[i] = (_Float16)(yssm[i] * g);
}

// ---------------------------------------------------------------------------
extern "C" void kernel_launch(void* const* d_in, const int* in_sizes, int n_in,
                              void* d_out, int out_size, void* d_ws, size_t ws_size,
                              hipStream_t stream) {
  (void)in_sizes; (void)n_in; (void)out_size; (void)ws_size;
  const float* x       = (const float*)d_in[0];
  const float* W_in    = (const float*)d_in[1];
  const float* conv_w  = (const float*)d_in[2];
  const float* conv_b  = (const float*)d_in[3];
  const float* W_xproj = (const float*)d_in[4];
  const float* W_dt    = (const float*)d_in[5];
  const float* b_dt    = (const float*)d_in[6];
  const float* A_log   = (const float*)d_in[7];
  const float* D_skip  = (const float*)d_in[8];
  const float* W_out   = (const float*)d_in[9];
  float* out = (float*)d_out;

  char* w = (char*)d_ws;
  auto carve = [&](size_t bytes) {
    char* p = w; w += (bytes + 255) & ~(size_t)255; return p;
  };
  _Float16* x16    = (_Float16*)carve((size_t)NTOK * DIM_ * 2);
  _Float16* WinT   = (_Float16*)carve((size_t)DIM_ * 2 * D_INNER * 2);  // [2E][512]
  _Float16* WxpT   = (_Float16*)carve((size_t)D_INNER * 64 * 2);        // [64][1024]
  _Float16* WdtT   = (_Float16*)carve((size_t)DT_RANK * D_INNER * 2);   // [1024][32]
  _Float16* WoutT  = (_Float16*)carve((size_t)D_INNER * DIM_ * 2);      // [512][1024]
  float*    xz     = (float*)   carve((size_t)NTOK * 2 * D_INNER * 4);
  float*    xh32   = (float*)   carve((size_t)NTOK * D_INNER * 4);
  _Float16* xh16   = (_Float16*)carve((size_t)NTOK * D_INNER * 2);
  float*    dbl    = (float*)   carve((size_t)NTOK * 64 * 4);
  _Float16* dtl16  = (_Float16*)carve((size_t)NTOK * DT_RANK * 2);
  float*    delta  = (float*)   carve((size_t)NTOK * D_INNER * 4);
  float*    yssm   = (float*)   carve((size_t)NTOK * D_INNER * 4);
  _Float16* yg16   = (_Float16*)carve((size_t)NTOK * D_INNER * 2);

  // activations: plain convert; weights: convert + transpose (once per call)
  cvt_f32_f16<<<(NTOK * DIM_ + 255) / 256, 256, 0, stream>>>(x, x16, NTOK * DIM_);
  auto cvtT = [&](const float* s, _Float16* d, int R, int Cc) {
    cvt_transpose_f16<<<(R * Cc + 255) / 256, 256, 0, stream>>>(s, d, R, Cc);
  };
  cvtT(W_in,    WinT,  DIM_,    2 * D_INNER);
  cvtT(W_xproj, WxpT,  D_INNER, 64);
  cvtT(W_dt,    WdtT,  DT_RANK, D_INNER);
  cvtT(W_out,   WoutT, D_INNER, DIM_);

  auto gemm = [&](const _Float16* A, const _Float16* BT, float* C,
                  const float* bias, int act, int M, int N, int K,
                  int lda, int ldc) {
    dim3 grid(N / 64, M / 128);
    wmma_gemm_f16f32<<<grid, 256, 0, stream>>>(A, BT, C, bias, act,
                                               M, N, K, lda, ldc);
  };

  // 1) xz = x @ W_in                     (8192x512 · 512x2048)
  gemm(x16, WinT, xz, nullptr, 0,
       NTOK, 2 * D_INNER, DIM_, DIM_, 2 * D_INNER);
  // 2) causal depthwise conv + SiLU
  conv_silu_kernel<<<(NTOK * D_INNER + 255) / 256, 256, 0, stream>>>(
      xz, conv_w, conv_b, xh32, xh16);
  // 3) dbl = xh @ W_xproj                (8192x1024 · 1024x64)
  gemm(xh16, WxpT, dbl, nullptr, 0,
       NTOK, 64, D_INNER, D_INNER, 64);
  // 4) delta = softplus(dtl @ W_dt + b_dt), fused epilogue (K=32 -> one WMMA step)
  dtl_cvt_kernel<<<(NTOK * DT_RANK + 255) / 256, 256, 0, stream>>>(dbl, dtl16);
  gemm(dtl16, WdtT, delta, b_dt, 1,
       NTOK, D_INNER, DT_RANK, DT_RANK, D_INNER);
  // 5) selective scan (sequential in L, parallel over 4096 channels)
  scan_kernel<<<64, 64, 0, stream>>>(delta, dbl, xh32, A_log, D_skip, yssm);
  // 6) gate with silu(z)
  gate_kernel<<<(NTOK * D_INNER + 255) / 256, 256, 0, stream>>>(yssm, xz, yg16);
  // 7) out = yg @ W_out                  (8192x1024 · 1024x512)
  gemm(yg16, WoutT, out, nullptr, 0,
       NTOK, DIM_, D_INNER, D_INNER, DIM_);
}
